// GNNLayer_81492709474503
// MI455X (gfx1250) — compile-verified, matrix operand order checked by reading, not run on hardware
//
#include <hip/hip_runtime.h>
#include <hip/hip_bf16.h>
#include <stdint.h>
#include <stddef.h>

// ---------------------------------------------------------------------------
// Types for CDNA5 WMMA (wave32, 16x16x32 bf16 -> f32)
// ---------------------------------------------------------------------------
typedef __attribute__((ext_vector_type(16))) __bf16          v16bf;
typedef __attribute__((ext_vector_type(16))) unsigned short  v16u;
typedef __attribute__((ext_vector_type(8)))  float           v8f;

// Each block = ONE wave32; each wave owns 2 M-tiles (32 rows) so every
// B-fragment fetched from L2 feeds two WMMAs (halves weight traffic).
#define ROWS_PER_WAVE 32
// LDS activation row stride: 264 ushorts = 528 B = 132 dwords -> lane stride
// of 4 banks, so 16-lane A-fragment ds_load_b128 sweeps all 64 LDS banks
// (256-wide rows would put every lane on bank 0).
#define ASTRIDE 264

// ---------------------------------------------------------------------------
// Weight-fragment arena layout (units: ushort / bf16 elements)
//   Each matrix (K x N) padded to Kp (mult 32) x Np (mult 16) and stored as
//   WMMA B-fragments: frag(kt,nt) = 32 lanes x 16 bf16, contiguous per lane.
// ---------------------------------------------------------------------------
enum : unsigned {
  F_MW0 = 0,                      // 32  x 256
  F_MW1 = F_MW0 + 32 * 256,       // 256 x 256
  F_MW2 = F_MW1 + 256 * 256,      // 256 x 128
  F_AW0 = F_MW2 + 256 * 128,      // 128 x 128
  F_AW1 = F_AW0 + 128 * 128,      // 128 x 128
  F_AW2 = F_AW1 + 128 * 128,      // 128 x 16 (only col 0 valid)
  F_VW0 = F_AW2 + 128 * 16,       // 128 x 256
  F_VW1 = F_VW0 + 128 * 256,      // 256 x 128
  F_UW0 = F_VW1 + 256 * 128,      // 128 x 256
  F_UW1 = F_UW0 + 128 * 256,      // 256 x 256
  F_UW2 = F_UW1 + 256 * 256,      // 256 x 16 (only col 0 valid)
  FRAG_TOTAL = F_UW2 + 256 * 16
};

// ---------------------------------------------------------------------------
// Helpers
// ---------------------------------------------------------------------------
__device__ __forceinline__ unsigned short f2bf(float f) {
  __bf16 h = (__bf16)f;                       // native v_cvt
  return __builtin_bit_cast(unsigned short, h);
}

// monotonic float<->uint encoding for atomicMax-based segment max
__device__ __forceinline__ unsigned encf(float f) {
  unsigned u = __float_as_uint(f);
  return (u & 0x80000000u) ? ~u : (u | 0x80000000u);
}
__device__ __forceinline__ float decf(unsigned u) {
  u = (u & 0x80000000u) ? (u & 0x7FFFFFFFu) : ~u;
  return __uint_as_float(u);
}

// A-fragment (16x32 bf16, MxK) from padded row-major LDS activation tile.
// Lane layout per ISA 7.12.2: lanes 0-15 row M=lane, K = [0..7],[16..23];
// lanes 16-31 row M=lane-16, K = [8..15],[24..31].  Two 16B DS loads/lane.
__device__ __forceinline__ v16bf load_a_frag(const unsigned short* act,
                                             int kbase, int lane) {
  const int lp = lane & 15, half = lane >> 4;
  const unsigned short* p = act + lp * ASTRIDE + kbase + 8 * half;
  v16u u;
#pragma unroll
  for (int i = 0; i < 8; ++i) { u[i] = p[i]; u[8 + i] = p[16 + i]; }
  return __builtin_bit_cast(v16bf, u);
}

// B-fragment: pre-swizzled in global workspace; one 32B load per lane.
__device__ __forceinline__ v16bf load_b_frag(const unsigned short* wf,
                                             int frag, int lane) {
  const unsigned short* p = wf + ((size_t)frag * 32 + lane) * 16;
  v16u u = *(const v16u*)p;
  return __builtin_bit_cast(v16bf, u);
}

__device__ __forceinline__ v8f wmma_bf16(v16bf a, v16bf b, v8f c) {
  return __builtin_amdgcn_wmma_f32_16x16x32_bf16(false, a, false, b,
                                                 (short)0, c, false, false);
}

// One fused MLP layer on a 32-row tile (2 M-tiles):
//   act_in (32 x K bf16 in LDS) @ W (K x N fragments) + bias -> act_out (32 x N)
// A-fragments register-cached for both M-tiles; each B-fragment loaded once
// and used by two WMMAs (independent accumulator chains -> XDL ILP).
template <int KTILES, bool RELU>
__device__ __forceinline__ void mlp32(const unsigned short* act_in,
                                      unsigned short* act_out, int N,
                                      const unsigned short* wfrag,
                                      const float* bias, int nvalid, int lane) {
  const int lp = lane & 15, half = lane >> 4;
  const int ntiles = N >> 4;

  v16bf a0[KTILES], a1[KTILES];
#pragma unroll
  for (int kt = 0; kt < KTILES; ++kt) {
    a0[kt] = load_a_frag(act_in, kt * 32, lane);
    a1[kt] = load_a_frag(act_in + 16 * ASTRIDE, kt * 32, lane);
  }

  for (int nt = 0; nt < ntiles; ++nt) {
    v8f c0 = {}, c1 = {};
#pragma unroll
    for (int kt = 0; kt < KTILES; ++kt) {
      v16bf b = load_b_frag(wfrag, kt * ntiles + nt, lane);
      c0 = wmma_bf16(a0[kt], b, c0);
      c1 = wmma_bf16(a1[kt], b, c1);
    }
    const int n = nt * 16 + lp;
    const float bb = (n < nvalid) ? bias[n] : 0.f;
#pragma unroll
    for (int i = 0; i < 8; ++i) {
      float x0 = c0[i] + bb, x1 = c1[i] + bb;
      if (RELU) { x0 = fmaxf(x0, 0.f); x1 = fmaxf(x1, 0.f); }
      const int m = i + 8 * half;
      act_out[m * ASTRIDE + n]        = f2bf(x0);
      act_out[(16 + m) * ASTRIDE + n] = f2bf(x1);
    }
  }
}

// Stage z = [nodes[s] | nodes[r] | edge] (padded to 32) into LDS (32 x 32)
__device__ __forceinline__ void stage_z(unsigned short* A, int e0, int E,
                                        const float* nodes, const float* edges,
                                        const int* senders, const int* receivers,
                                        int lane) {
  for (int t = lane; t < ROWS_PER_WAVE * 32; t += 32) {
    const int m = t >> 5, k = t & 31;
    const int e = e0 + m;
    float val = 0.f;
    if (e < E) {
      if (k < 3)       val = nodes[(size_t)senders[e] * 3 + k];
      else if (k < 6)  val = nodes[(size_t)receivers[e] * 3 + (k - 3)];
      else if (k < 10) val = edges[(size_t)e * 4 + (k - 6)];
    }
    A[m * ASTRIDE + k] = f2bf(val);
  }
}

// ---------------------------------------------------------------------------
// Kernel 0: convert one fp32 weight matrix into bf16 WMMA B-fragments
// ---------------------------------------------------------------------------
__global__ void prep_frag(const float* __restrict__ W, int K, int N,
                          int Kp, int Np, unsigned short* __restrict__ dst) {
  const int t = blockIdx.x * blockDim.x + threadIdx.x;
  if (t >= Kp * Np) return;
  const int ntiles = Np >> 4;
  const int frag = t >> 9, r = t & 511;
  const int lane = r >> 4, j = r & 15;
  const int kt = frag / ntiles, nt = frag % ntiles;
  const int half = lane >> 4, lp = lane & 15;
  const int v = j >> 1, e = j & 1;
  const int kl = (v < 4) ? (8 * half + 2 * v + e)
                         : (16 + 8 * half + 2 * (v - 4) + e);
  const int k = kt * 32 + kl, n = nt * 16 + lp;
  const float val = (k < K && n < N) ? W[(size_t)k * N + n] : 0.f;
  dst[t] = f2bf(val);
}

// ---------------------------------------------------------------------------
// Kernel 1: init segment-max / segment-sum / aggregation buffers
// ---------------------------------------------------------------------------
__global__ void init_ws(unsigned* __restrict__ m_bits, float* __restrict__ ssum,
                        float* __restrict__ aggr, int n_node) {
  const int i = blockIdx.x * blockDim.x + threadIdx.x;
  if (i < n_node * 128) aggr[i] = 0.f;
  if (i < n_node) { m_bits[i] = encf(-1e9f); ssum[i] = 0.f; }
}

// ---------------------------------------------------------------------------
// Kernel 2: edge pass 1 — psi1 + psi2 -> attention logits w; segment max
// ---------------------------------------------------------------------------
__global__ __launch_bounds__(32)
void edge_pass1(const float* __restrict__ nodes, const float* __restrict__ edges,
                const int* __restrict__ senders, const int* __restrict__ receivers,
                const unsigned short* __restrict__ frags,
                const float* mb0, const float* mb1, const float* mb2,
                const float* ab0, const float* ab1, const float* ab2,
                float* __restrict__ w_out, unsigned* __restrict__ m_bits, int E) {
  __shared__ __align__(16) unsigned short sA[ROWS_PER_WAVE * ASTRIDE];
  __shared__ __align__(16) unsigned short sB[ROWS_PER_WAVE * ASTRIDE];
  const int lane = threadIdx.x & 31;
  const int lp = lane & 15, half = lane >> 4;
  const int e0 = blockIdx.x * ROWS_PER_WAVE;
  unsigned short* A = sA;
  unsigned short* B = sB;

  stage_z(A, e0, E, nodes, edges, senders, receivers, lane);

  mlp32<1, true>(A, B, 256, frags + F_MW0, mb0, 256, lane);   // psi1.0  32->256
  mlp32<8, true>(B, A, 256, frags + F_MW1, mb1, 256, lane);   // psi1.1 256->256
  mlp32<8, true>(A, B, 128, frags + F_MW2, mb2, 128, lane);   // psi1.2 256->128 (q)
  mlp32<4, true>(B, A, 128, frags + F_AW0, ab0, 128, lane);   // psi2.0 128->128
  mlp32<4, true>(A, B, 128, frags + F_AW1, ab1, 128, lane);   // psi2.1 128->128

  // psi2.2: 128 -> 1 (padded to 16), inline; only column 0 is valid
  v8f c0 = {}, c1 = {};
#pragma unroll
  for (int kt = 0; kt < 4; ++kt) {
    v16bf b = load_b_frag(frags + F_AW2, kt, lane);
    c0 = wmma_bf16(load_a_frag(B, kt * 32, lane), b, c0);
    c1 = wmma_bf16(load_a_frag(B + 16 * ASTRIDE, kt * 32, lane), b, c1);
  }
  const float b0 = ab2[0];
  if (lp == 0) {
#pragma unroll
    for (int i = 0; i < 8; ++i) {
      const int m = i + 8 * half;
      const int ea = e0 + m, eb = e0 + 16 + m;
      if (ea < E) {
        const float wv = c0[i] + b0;
        w_out[ea] = wv;
        atomicMax(&m_bits[receivers[ea]], encf(wv));
      }
      if (eb < E) {
        const float wv = c1[i] + b0;
        w_out[eb] = wv;
        atomicMax(&m_bits[receivers[eb]], encf(wv));
      }
    }
  }
}

// ---------------------------------------------------------------------------
// Kernel 3: scalar softmax pass — ex = exp(w - m[r]); s[r] += ex
// ---------------------------------------------------------------------------
__global__ void seg_softmax(const float* __restrict__ w,
                            const unsigned* __restrict__ m_bits,
                            const int* __restrict__ receivers,
                            float* __restrict__ ex, float* __restrict__ ssum, int E) {
  const int e = blockIdx.x * blockDim.x + threadIdx.x;
  if (e >= E) return;
  const int r = receivers[e];
  float m = decf(m_bits[r]);
  m = fmaxf(m, -1e9f);
  const float x = __expf(w[e] - m);
  ex[e] = x;
  atomicAdd(&ssum[r], x);
}

// ---------------------------------------------------------------------------
// Kernel 4: edge pass 2 — recompute psi1 -> q, psi3 -> v, attn-weighted scatter
// ---------------------------------------------------------------------------
__global__ __launch_bounds__(32)
void edge_pass2(const float* __restrict__ nodes, const float* __restrict__ edges,
                const int* __restrict__ senders, const int* __restrict__ receivers,
                const unsigned short* __restrict__ frags,
                const float* mb0, const float* mb1, const float* mb2,
                const float* vb0, const float* vb1,
                const float* __restrict__ ex, const float* __restrict__ ssum,
                float* __restrict__ aggr, int E) {
  __shared__ __align__(16) unsigned short sA[ROWS_PER_WAVE * ASTRIDE];
  __shared__ __align__(16) unsigned short sB[ROWS_PER_WAVE * ASTRIDE];
  const int lane = threadIdx.x & 31;
  const int lp = lane & 15, half = lane >> 4;
  const int e0 = blockIdx.x * ROWS_PER_WAVE;
  unsigned short* A = sA;
  unsigned short* B = sB;

  stage_z(A, e0, E, nodes, edges, senders, receivers, lane);

  mlp32<1, true>(A, B, 256, frags + F_MW0, mb0, 256, lane);   // psi1.0
  mlp32<8, true>(B, A, 256, frags + F_MW1, mb1, 256, lane);   // psi1.1
  mlp32<8, true>(A, B, 128, frags + F_MW2, mb2, 128, lane);   // psi1.2 -> q
  mlp32<4, true>(B, A, 256, frags + F_VW0, vb0, 256, lane);   // psi3.0 128->256

  // per-lane attention scale for the 16 rows this lane contributes to
  float at0[8], at1[8]; int rc0[8], rc1[8];
#pragma unroll
  for (int i = 0; i < 8; ++i) {
    const int m = i + 8 * half;
    const int ea = e0 + m, eb = e0 + 16 + m;
    if (ea < E) { rc0[i] = receivers[ea]; at0[i] = ex[ea] / (ssum[rc0[i]] + 1e-12f); }
    else        { rc0[i] = 0; at0[i] = 0.f; }
    if (eb < E) { rc1[i] = receivers[eb]; at1[i] = ex[eb] / (ssum[rc1[i]] + 1e-12f); }
    else        { rc1[i] = 0; at1[i] = 0.f; }
  }

  // psi3.1 inline: 256 -> 128, ReLU, scale by attn, scatter-add to aggr.
  v16bf a0[8], a1[8];
#pragma unroll
  for (int kt = 0; kt < 8; ++kt) {
    a0[kt] = load_a_frag(A, kt * 32, lane);
    a1[kt] = load_a_frag(A + 16 * ASTRIDE, kt * 32, lane);
  }
  for (int nt = 0; nt < 8; ++nt) {
    v8f c0 = {}, c1 = {};
#pragma unroll
    for (int kt = 0; kt < 8; ++kt) {
      v16bf b = load_b_frag(frags + F_VW1, kt * 8 + nt, lane);
      c0 = wmma_bf16(a0[kt], b, c0);
      c1 = wmma_bf16(a1[kt], b, c1);
    }
    const float bb = vb1[nt * 16 + lp];
    const int col = nt * 16 + lp;
#pragma unroll
    for (int i = 0; i < 8; ++i) {
      const int m = i + 8 * half;
      if (e0 + m < E) {
        const float x = fmaxf(c0[i] + bb, 0.f);
        atomicAdd(&aggr[(size_t)rc0[i] * 128 + col], x * at0[i]);
      }
      if (e0 + 16 + m < E) {
        const float x = fmaxf(c1[i] + bb, 0.f);
        atomicAdd(&aggr[(size_t)rc1[i] * 128 + col], x * at1[i]);
      }
    }
  }
}

// ---------------------------------------------------------------------------
// Kernel 5: node pass — psi4 decoder on aggregated features
// ---------------------------------------------------------------------------
__global__ __launch_bounds__(32)
void node_pass(const float* __restrict__ aggr,
               const unsigned short* __restrict__ frags,
               const float* ub0, const float* ub1, const float* ub2,
               float* __restrict__ out, int n_node) {
  __shared__ __align__(16) unsigned short sA[ROWS_PER_WAVE * ASTRIDE];
  __shared__ __align__(16) unsigned short sB[ROWS_PER_WAVE * ASTRIDE];
  const int lane = threadIdx.x & 31;
  const int lp = lane & 15, half = lane >> 4;
  const int n0 = blockIdx.x * ROWS_PER_WAVE;
  unsigned short* A = sA;
  unsigned short* B = sB;

  // stage aggr tile (32 x 128) -> bf16 LDS
  for (int t = lane; t < ROWS_PER_WAVE * 128; t += 32) {
    const int m = t >> 7, k = t & 127;
    const int n = n0 + m;
    const float val = (n < n_node) ? aggr[(size_t)n * 128 + k] : 0.f;
    A[m * ASTRIDE + k] = f2bf(val);
  }

  mlp32<4, true>(A, B, 256, frags + F_UW0, ub0, 256, lane);   // psi4.0 128->256
  mlp32<8, true>(B, A, 256, frags + F_UW1, ub1, 256, lane);   // psi4.1 256->256

  // psi4.2 inline: 256 -> 1 (padded to 16)
  v8f c0 = {}, c1 = {};
#pragma unroll
  for (int kt = 0; kt < 8; ++kt) {
    v16bf b = load_b_frag(frags + F_UW2, kt, lane);
    c0 = wmma_bf16(load_a_frag(A, kt * 32, lane), b, c0);
    c1 = wmma_bf16(load_a_frag(A + 16 * ASTRIDE, kt * 32, lane), b, c1);
  }
  const float b0 = ub2[0];
  if (lp == 0) {
#pragma unroll
    for (int i = 0; i < 8; ++i) {
      const int m = i + 8 * half;
      if (n0 + m < n_node)      out[n0 + m]      = c0[i] + b0;
      if (n0 + 16 + m < n_node) out[n0 + 16 + m] = c1[i] + b0;
    }
  }
}

// ---------------------------------------------------------------------------
// Host launcher
// ---------------------------------------------------------------------------
extern "C" void kernel_launch(void* const* d_in, const int* in_sizes, int n_in,
                              void* d_out, int out_size, void* d_ws, size_t ws_size,
                              hipStream_t stream) {
  const float* nodes     = (const float*)d_in[0];
  const float* edges     = (const float*)d_in[1];
  const int*   senders   = (const int*)d_in[2];
  const int*   receivers = (const int*)d_in[3];
  const float* mw0 = (const float*)d_in[4];  const float* mb0 = (const float*)d_in[5];
  const float* mw1 = (const float*)d_in[6];  const float* mb1 = (const float*)d_in[7];
  const float* mw2 = (const float*)d_in[8];  const float* mb2 = (const float*)d_in[9];
  const float* aw0 = (const float*)d_in[10]; const float* ab0 = (const float*)d_in[11];
  const float* aw1 = (const float*)d_in[12]; const float* ab1 = (const float*)d_in[13];
  const float* aw2 = (const float*)d_in[14]; const float* ab2 = (const float*)d_in[15];
  const float* vw0 = (const float*)d_in[16]; const float* vb0 = (const float*)d_in[17];
  const float* vw1 = (const float*)d_in[18]; const float* vb1 = (const float*)d_in[19];
  const float* uw0 = (const float*)d_in[20]; const float* ub0 = (const float*)d_in[21];
  const float* uw1 = (const float*)d_in[22]; const float* ub1 = (const float*)d_in[23];
  const float* uw2 = (const float*)d_in[24]; const float* ub2 = (const float*)d_in[25];

  const int E  = in_sizes[2];        // senders length
  const int Nn = in_sizes[0] / 3;    // node count

  // workspace carve-up (~33 MB total)
  char* ws = (char*)d_ws;
  unsigned short* frags = (unsigned short*)ws;
  size_t off = (size_t)FRAG_TOTAL * sizeof(unsigned short);
  float*    w_buf  = (float*)(ws + off);    off += (size_t)E * 4;
  float*    ex_buf = (float*)(ws + off);    off += (size_t)E * 4;
  unsigned* m_bits = (unsigned*)(ws + off); off += (size_t)Nn * 4;
  float*    s_buf  = (float*)(ws + off);    off += (size_t)Nn * 4;
  float*    aggr   = (float*)(ws + off);    off += (size_t)Nn * 128 * 4;
  (void)ws_size; (void)n_in; (void)out_size;

  // 1) weight fragments (bf16, WMMA B layout) — deterministic each call
  struct PrepArgs { const float* W; int K, N, Kp, Np; unsigned foff; };
  const PrepArgs preps[11] = {
    { mw0,  10, 256,  32, 256, F_MW0 },
    { mw1, 256, 256, 256, 256, F_MW1 },
    { mw2, 256, 128, 256, 128, F_MW2 },
    { aw0, 128, 128, 128, 128, F_AW0 },
    { aw1, 128, 128, 128, 128, F_AW1 },
    { aw2, 128,   1, 128,  16, F_AW2 },
    { vw0, 128, 256, 128, 256, F_VW0 },
    { vw1, 256, 128, 256, 128, F_VW1 },
    { uw0, 128, 256, 128, 256, F_UW0 },
    { uw1, 256, 256, 256, 256, F_UW1 },
    { uw2, 256,   1, 256,  16, F_UW2 },
  };
  for (int i = 0; i < 11; ++i) {
    const int total = preps[i].Kp * preps[i].Np;
    prep_frag<<<(total + 255) / 256, 256, 0, stream>>>(
        preps[i].W, preps[i].K, preps[i].N, preps[i].Kp, preps[i].Np,
        frags + preps[i].foff);
  }

  // 2) init m/s/aggr
  {
    const int total = Nn * 128;
    init_ws<<<(total + 255) / 256, 256, 0, stream>>>(m_bits, s_buf, aggr, Nn);
  }

  const int edgeBlocks = (E + ROWS_PER_WAVE - 1) / ROWS_PER_WAVE;
  const int nodeBlocks = (Nn + ROWS_PER_WAVE - 1) / ROWS_PER_WAVE;
  const dim3 blk(32);

  // 3) edge pass 1: logits + segment max
  edge_pass1<<<edgeBlocks, blk, 0, stream>>>(
      nodes, edges, senders, receivers, frags,
      mb0, mb1, mb2, ab0, ab1, ab2, w_buf, m_bits, E);

  // 4) softmax numerators + segment sum
  seg_softmax<<<(E + 255) / 256, 256, 0, stream>>>(
      w_buf, m_bits, receivers, ex_buf, s_buf, E);

  // 5) edge pass 2: values + attention-weighted aggregation
  edge_pass2<<<edgeBlocks, blk, 0, stream>>>(
      nodes, edges, senders, receivers, frags,
      mb0, mb1, mb2, vb0, vb1, ex_buf, s_buf, aggr, E);

  // 6) node decoder
  node_pass<<<nodeBlocks, blk, 0, stream>>>(
      aggr, frags, ub0, ub1, ub2, (float*)d_out, Nn);
}
